// GroupedQueryAttention_65687229825704
// MI455X (gfx1250) — matrix operand order repaired and checked by reading, not compiled, not run on hardware
//
#include <hip/hip_runtime.h>
#include <hip/hip_bf16.h>

typedef __bf16 bf16;
typedef __attribute__((ext_vector_type(16))) __bf16 v16bf;
typedef __attribute__((ext_vector_type(8)))  __bf16 v8bf;
typedef __attribute__((ext_vector_type(8)))  float  v8f;

// ---------------- problem constants ----------------
constexpr int Bq   = 2;
constexpr int S    = 2048;
constexpr int DM   = 2048;
constexpr int NH   = 32;
constexpr int NKV  = 8;
constexpr int HD   = 64;
constexpr int MR   = Bq * S;          // 4096 rows of x
constexpr float SCALE = 0.125f;       // 1/sqrt(64)

// d_out layout (fp32): [out | k_cache | v_cache]
constexpr size_t OUT_K_OFF = (size_t)MR * DM;                   // 8388608
constexpr size_t OUT_V_OFF = OUT_K_OFF + (size_t)MR * NKV * HD; // 10485760

// ---------------- workspace layout (bytes) ----------------
constexpr size_t OFF_XBF   = 0;                                  // 16 MB
constexpr size_t OFF_WQBF  = OFF_XBF  + (size_t)MR*DM*2;         //  8 MB
constexpr size_t OFF_WKBF  = OFF_WQBF + (size_t)DM*DM*2;         //  2 MB
constexpr size_t OFF_WVBF  = OFF_WKBF + (size_t)(NKV*HD)*DM*2;   //  2 MB
constexpr size_t OFF_WOBF  = OFF_WVBF + (size_t)(NKV*HD)*DM*2;   //  8 MB
constexpr size_t OFF_QF32  = OFF_WOBF + (size_t)DM*DM*2;         // 32 MB (aliased by AO bf16 later)
constexpr size_t OFF_KF32  = OFF_QF32 + (size_t)MR*DM*4;         //  8 MB
constexpr size_t OFF_VF32  = OFF_KF32 + (size_t)MR*NKV*HD*4;     //  8 MB
constexpr size_t OFF_QBF   = OFF_VF32 + (size_t)MR*NKV*HD*4;     // 16 MB
constexpr size_t OFF_KBF   = OFF_QBF  + (size_t)MR*DM*2;         //  4 MB
constexpr size_t OFF_VTBF  = OFF_KBF  + (size_t)MR*NKV*HD*2;     //  4 MB
constexpr size_t OFF_AOBF  = OFF_QF32;   // alias: q_f32 dead after rope_q

// ---------------- CDNA5 async global->LDS helpers ----------------
__device__ __forceinline__ void async_ld_b128(unsigned lds_off, const void* gaddr) {
  // GLOBAL_LOAD_ASYNC_TO_LDS_B128, GV mode (saddr=null); tracked by ASYNCcnt
  asm volatile("global_load_async_to_lds_b128 %0, %1, off"
               :: "v"(lds_off), "v"(gaddr) : "memory");
}
__device__ __forceinline__ void wait_async_4() {
  asm volatile("s_wait_asynccnt 0x4" ::: "memory");
}
__device__ __forceinline__ void wait_async_0() {
  asm volatile("s_wait_asynccnt 0x0" ::: "memory");
}
__device__ __forceinline__ void wait_ds_0() {
  asm volatile("s_wait_dscnt 0x0" ::: "memory");
}

// ---------------- fragment loaders (contiguous k order k = 16*(lane/16) + s, A and B alike) ----------------
// A-matrix 16x32 bf16: lane<16 row=lane K=0..15; lane>=16 row=lane-16 K=16..31
__device__ __forceinline__ v16bf load_frag_a(const bf16* base, int r0, int k0, int lda) {
  int lane = threadIdx.x & 31;
  int r = lane & 15, h = lane >> 4;
  const bf16* p = base + (size_t)(r0 + r) * lda + k0 + h * 16;
  union { v16bf v; v8bf h2[2]; } u;
  u.h2[0] = *(const v8bf*)(p);
  u.h2[1] = *(const v8bf*)(p + 8);
  return u.v;
}
// B-matrix 32x16 bf16 from row-major [n][k] source: lane<16 n=lane K=0..15; lane>=16 K=16..31
__device__ __forceinline__ v16bf load_frag_b(const bf16* base, int n0, int k0, int lda) {
  int lane = threadIdx.x & 31;
  int r = lane & 15, h = lane >> 4;
  const bf16* p = base + (size_t)(n0 + r) * lda + k0 + h * 16;
  return *(const v16bf*)p;
}

// ---------------- fp32 -> bf16 convert ----------------
__global__ void gqa_cvt_bf16(const float* __restrict__ in, bf16* __restrict__ out, int n) {
  int i = blockIdx.x * blockDim.x + threadIdx.x;
  if (i < n) out[i] = (bf16)in[i];
}

// ---------------- tiled bf16 GEMM: C[m,n] = sum_k A[m,k]*Bt[n,k] ----------------
// 64x64 block tile, K-step 32, 4 waves; double-buffered LDS fed by async DMA
__global__ __launch_bounds__(128) void gqa_gemm_bf16(
    const bf16* __restrict__ A,   // M x K row-major
    const bf16* __restrict__ Bt,  // N x K row-major
    float* __restrict__ C,        // M x N
    int M, int N, int K) {
  __shared__ bf16 As[2][64 * 32];
  __shared__ bf16 Bs[2][64 * 32];
  const int tid  = threadIdx.x;
  const int wave = tid >> 5;
  const int m0 = blockIdx.y * 64;
  const int n0 = blockIdx.x * 64;

  // 4 KB tile = 256 x 16B chunks; thread t handles chunks t and t+128.
  // chunk c: row = c>>2 (four 16B pieces per 64B row), elem = (c&3)*8
  const int r0c = tid >> 2,         e0 = (tid & 3) * 8;
  const int r1c = (tid + 128) >> 2, e1 = ((tid + 128) & 3) * 8;
  const bf16* Ag0 = A  + (size_t)(m0 + r0c) * K + e0;
  const bf16* Ag1 = A  + (size_t)(m0 + r1c) * K + e1;
  const bf16* Bg0 = Bt + (size_t)(n0 + r0c) * K + e0;
  const bf16* Bg1 = Bt + (size_t)(n0 + r1c) * K + e1;
  unsigned la0[2], la1[2], lb0[2], lb1[2];
#pragma unroll
  for (int p = 0; p < 2; ++p) {
    la0[p] = (unsigned)(size_t)&As[p][r0c * 32 + e0];
    la1[p] = (unsigned)(size_t)&As[p][r1c * 32 + e1];
    lb0[p] = (unsigned)(size_t)&Bs[p][r0c * 32 + e0];
    lb1[p] = (unsigned)(size_t)&Bs[p][r1c * 32 + e1];
  }

  v8f acc[4] = {v8f{}, v8f{}, v8f{}, v8f{}};

  // prime buffer 0 with the k=0 tile (4 async ops per thread -> ASYNCcnt += 4/wave)
  async_ld_b128(la0[0], Ag0);
  async_ld_b128(la1[0], Ag1);
  async_ld_b128(lb0[0], Bg0);
  async_ld_b128(lb1[0], Bg1);

  int buf = 0;
  for (int k0 = 0; k0 < K; k0 += 32) {
    const int kn = k0 + 32;
    if (kn < K) {            // pipeline: issue next tile into the other buffer
      async_ld_b128(la0[buf ^ 1], Ag0 + kn);
      async_ld_b128(la1[buf ^ 1], Ag1 + kn);
      async_ld_b128(lb0[buf ^ 1], Bg0 + kn);
      async_ld_b128(lb1[buf ^ 1], Bg1 + kn);
      wait_async_4();        // current tile landed; next may stay in flight
    } else {
      wait_async_0();
    }
    __syncthreads();
    v16bf af = load_frag_a(As[buf], wave * 16, 0, 32);
#pragma unroll
    for (int nt = 0; nt < 4; ++nt) {
      v16bf bfrag = load_frag_b(Bs[buf], nt * 16, 0, 32);
      acc[nt] = __builtin_amdgcn_wmma_f32_16x16x32_bf16(
          false, af, false, bfrag, (short)0, acc[nt], false, false);
    }
    __syncthreads();         // all waves done reading before buf is overwritten
    buf ^= 1;
  }

  const int lane = tid & 31, r = lane & 15, h = lane >> 4;
#pragma unroll
  for (int nt = 0; nt < 4; ++nt)
#pragma unroll
    for (int j = 0; j < 8; ++j) {
      int m = m0 + wave * 16 + j + 8 * h;
      int n = n0 + nt * 16 + r;
      C[(size_t)m * N + n] = acc[nt][j];
    }
}

// ---------------- RoPE on Q: q_f32[m][h*64+d] -> q_bf16[b][h][s][d] ----------------
__global__ void gqa_rope_q(const float* __restrict__ qf, const float* __restrict__ cosb,
                           const float* __restrict__ sinb, bf16* __restrict__ qb) {
  int idx = blockIdx.x * blockDim.x + threadIdx.x;   // MR*NH*32
  if (idx >= MR * NH * (HD / 2)) return;
  int i = idx & 31, h = (idx >> 5) & 31, m = idx >> 10;
  int s = m & (S - 1), b = m >> 11;
  const float* p = qf + (size_t)m * DM + h * HD + 2 * i;
  float c = cosb[s * 32 + i], sn = sinb[s * 32 + i];
  float x0 = p[0], x1 = p[1];
  size_t o = (((size_t)(b * NH + h) * S) + s) * HD + 2 * i;
  qb[o]     = (bf16)(x0 * c - x1 * sn);
  qb[o + 1] = (bf16)(x0 * sn + x1 * c);
}

// ---------------- RoPE on K: k_f32 -> k_bf16[b][g][s][d] + fp32 k_cache ----------------
__global__ void gqa_rope_k(const float* __restrict__ kf, const float* __restrict__ cosb,
                           const float* __restrict__ sinb, bf16* __restrict__ kb,
                           float* __restrict__ out) {
  int idx = blockIdx.x * blockDim.x + threadIdx.x;   // MR*NKV*32
  if (idx >= MR * NKV * (HD / 2)) return;
  int i = idx & 31, g = (idx >> 5) & 7, m = idx >> 8;
  int s = m & (S - 1), b = m >> 11;
  const float* p = kf + (size_t)m * (NKV * HD) + g * HD + 2 * i;
  float c = cosb[s * 32 + i], sn = sinb[s * 32 + i];
  float x0 = p[0], x1 = p[1];
  float o0 = x0 * c - x1 * sn, o1 = x0 * sn + x1 * c;
  size_t co = OUT_K_OFF + (size_t)m * (NKV * HD) + g * HD + 2 * i;
  out[co] = o0; out[co + 1] = o1;
  size_t o = (((size_t)(b * NKV + g) * S) + s) * HD + 2 * i;
  kb[o] = (bf16)o0; kb[o + 1] = (bf16)o1;
}

// ---------------- V: fp32 cache + transposed bf16 [b][g][d][s] ----------------
__global__ void gqa_pack_v(const float* __restrict__ vf, bf16* __restrict__ vt,
                           float* __restrict__ out) {
  int idx = blockIdx.x * blockDim.x + threadIdx.x;   // MR*NKV*HD
  if (idx >= MR * NKV * HD) return;
  int d = idx & 63, g = (idx >> 6) & 7, m = idx >> 9;
  int s = m & (S - 1), b = m >> 11;
  float v = vf[(size_t)m * (NKV * HD) + g * HD + d];
  out[OUT_V_OFF + (size_t)m * (NKV * HD) + g * HD + d] = v;
  vt[((size_t)(b * NKV + g) * HD + d) * S + s] = (bf16)v;
}

// ---------------- flash attention: one wave per (b, head, 16-query block) ----------------
__global__ __launch_bounds__(32) void gqa_flash(
    const bf16* __restrict__ Q,   // [B][NH][S][HD]
    const bf16* __restrict__ Kt,  // [B][NKV][S][HD]
    const bf16* __restrict__ Vt,  // [B][NKV][HD][S]
    bf16* __restrict__ O) {       // [B][S][NH*HD]
  const int qblk = blockIdx.x, h = blockIdx.y, b = blockIdx.z;
  const int g = h >> 2;                 // N_REP = 4
  const int q0 = qblk * 16;
  const bf16* Qp = Q  + ((size_t)(b * NH  + h) * S)  * HD;
  const bf16* Kp = Kt + ((size_t)(b * NKV + g) * S)  * HD;
  const bf16* Vp = Vt + ((size_t)(b * NKV + g) * HD) * S;

  const int lane = threadIdx.x & 31, r = lane & 15, hh = lane >> 4;

  const v16bf qa0 = load_frag_a(Qp, q0, 0, HD);
  const v16bf qa1 = load_frag_a(Qp, q0, 32, HD);

  float m_run[8], l_run[8];
#pragma unroll
  for (int j = 0; j < 8; ++j) { m_run[j] = -3.0e38f; l_run[j] = 0.f; }
  v8f accO[4] = {v8f{}, v8f{}, v8f{}, v8f{}};

  __shared__ bf16 Plds[16 * 32];

  const int nkb = (q0 + 47) >> 5;       // 32-key blocks covering causal range
  for (int kb = 0; kb < nkb; ++kb) {
    const int kbase = kb * 32;
    if (kbase + 32 < S) {   // prefetch next key block of K and V (global_prefetch_b8)
      __builtin_prefetch(Kp + (size_t)(kbase + 32 + lane) * HD, 0, 0);
      __builtin_prefetch(Vp + (size_t)lane * S + kbase + 32, 0, 0);
      __builtin_prefetch(Vp + (size_t)(32 + lane) * S + kbase + 32, 0, 0);
    }
    v8f sc0 = v8f{}, sc1 = v8f{};
    {
      v16bf bk = load_frag_b(Kp, kbase, 0, HD);
      sc0 = __builtin_amdgcn_wmma_f32_16x16x32_bf16(false, qa0, false, bk, (short)0, sc0, false, false);
      bk = load_frag_b(Kp, kbase, 32, HD);
      sc0 = __builtin_amdgcn_wmma_f32_16x16x32_bf16(false, qa1, false, bk, (short)0, sc0, false, false);
      bk = load_frag_b(Kp, kbase + 16, 0, HD);
      sc1 = __builtin_amdgcn_wmma_f32_16x16x32_bf16(false, qa0, false, bk, (short)0, sc1, false, false);
      bk = load_frag_b(Kp, kbase + 16, 32, HD);
      sc1 = __builtin_amdgcn_wmma_f32_16x16x32_bf16(false, qa1, false, bk, (short)0, sc1, false, false);
    }
    float sv0[8], sv1[8], mnew[8];
#pragma unroll
    for (int j = 0; j < 8; ++j) {
      int row = q0 + j + 8 * hh;
      float v0 = (kbase + r      <= row) ? sc0[j] * SCALE : -1.0e30f;
      float v1 = (kbase + 16 + r <= row) ? sc1[j] * SCALE : -1.0e30f;
      sv0[j] = v0; sv1[j] = v1;
      float mx = fmaxf(v0, v1);
      mx = fmaxf(mx, __shfl_xor(mx, 1, 32));
      mx = fmaxf(mx, __shfl_xor(mx, 2, 32));
      mx = fmaxf(mx, __shfl_xor(mx, 4, 32));
      mx = fmaxf(mx, __shfl_xor(mx, 8, 32));
      mnew[j] = fmaxf(m_run[j], mx);
    }
#pragma unroll
    for (int j = 0; j < 8; ++j) {
      float p0 = __expf(sv0[j] - mnew[j]);
      float p1 = __expf(sv1[j] - mnew[j]);
      float rs = p0 + p1;
      rs += __shfl_xor(rs, 1, 32);
      rs += __shfl_xor(rs, 2, 32);
      rs += __shfl_xor(rs, 4, 32);
      rs += __shfl_xor(rs, 8, 32);
      float corr = __expf(m_run[j] - mnew[j]);
      l_run[j] = l_run[j] * corr + rs;
      m_run[j] = mnew[j];
      accO[0][j] *= corr; accO[1][j] *= corr; accO[2][j] *= corr; accO[3][j] *= corr;
      Plds[(j + 8 * hh) * 32 + r]      = (bf16)p0;
      Plds[(j + 8 * hh) * 32 + 16 + r] = (bf16)p1;
    }
    // single-wave workgroup: same-wave DS ops are in-order; just drain DScnt
    wait_ds_0();
    v16bf pa = load_frag_a(Plds, 0, 0, 32);
#pragma unroll
    for (int nt = 0; nt < 4; ++nt) {
      v16bf vb = load_frag_b(Vp, nt * 16, kbase, S);
      accO[nt] = __builtin_amdgcn_wmma_f32_16x16x32_bf16(
          false, pa, false, vb, (short)0, accO[nt], false, false);
    }
    wait_ds_0();   // Plds reads done before next iteration's writes
  }
#pragma unroll
  for (int nt = 0; nt < 4; ++nt)
#pragma unroll
    for (int j = 0; j < 8; ++j) {
      int row = j + 8 * hh;
      size_t o = ((size_t)(b * S + q0 + row)) * DM + h * HD + nt * 16 + r;
      O[o] = (bf16)(accO[nt][j] / l_run[j]);
    }
}

// ---------------- launch ----------------
extern "C" void kernel_launch(void* const* d_in, const int* in_sizes, int n_in,
                              void* d_out, int out_size, void* d_ws, size_t ws_size,
                              hipStream_t stream) {
  const float* x    = (const float*)d_in[0];
  const float* wq   = (const float*)d_in[1];
  const float* wk   = (const float*)d_in[2];
  const float* wv   = (const float*)d_in[3];
  const float* wo   = (const float*)d_in[4];
  const float* cosb = (const float*)d_in[5];
  const float* sinb = (const float*)d_in[6];
  float* out = (float*)d_out;
  char*  ws  = (char*)d_ws;

  bf16* x_bf  = (bf16*)(ws + OFF_XBF);
  bf16* wq_bf = (bf16*)(ws + OFF_WQBF);
  bf16* wk_bf = (bf16*)(ws + OFF_WKBF);
  bf16* wv_bf = (bf16*)(ws + OFF_WVBF);
  bf16* wo_bf = (bf16*)(ws + OFF_WOBF);
  float* q_f  = (float*)(ws + OFF_QF32);
  float* k_f  = (float*)(ws + OFF_KF32);
  float* v_f  = (float*)(ws + OFF_VF32);
  bf16* q_bf  = (bf16*)(ws + OFF_QBF);
  bf16* k_bf  = (bf16*)(ws + OFF_KBF);
  bf16* vt_bf = (bf16*)(ws + OFF_VTBF);
  bf16* ao_bf = (bf16*)(ws + OFF_AOBF);

  auto cvt = [&](const float* src, bf16* dst, int n) {
    gqa_cvt_bf16<<<(n + 255) / 256, 256, 0, stream>>>(src, dst, n);
  };
  cvt(x,  x_bf,  MR * DM);
  cvt(wq, wq_bf, DM * DM);
  cvt(wk, wk_bf, NKV * HD * DM);
  cvt(wv, wv_bf, NKV * HD * DM);
  cvt(wo, wo_bf, DM * DM);

  // projections
  gqa_gemm_bf16<<<dim3(DM / 64, MR / 64), 128, 0, stream>>>(x_bf, wq_bf, q_f, MR, DM, DM);
  gqa_gemm_bf16<<<dim3((NKV * HD) / 64, MR / 64), 128, 0, stream>>>(x_bf, wk_bf, k_f, MR, NKV * HD, DM);
  gqa_gemm_bf16<<<dim3((NKV * HD) / 64, MR / 64), 128, 0, stream>>>(x_bf, wv_bf, v_f, MR, NKV * HD, DM);

  // rope + repack (+ caches into d_out)
  gqa_rope_q<<<(MR * NH * 32 + 255) / 256, 256, 0, stream>>>(q_f, cosb, sinb, q_bf);
  gqa_rope_k<<<(MR * NKV * 32 + 255) / 256, 256, 0, stream>>>(k_f, cosb, sinb, k_bf, out);
  gqa_pack_v<<<(MR * NKV * HD + 255) / 256, 256, 0, stream>>>(v_f, vt_bf, out);

  // attention
  gqa_flash<<<dim3(S / 16, NH, Bq), 32, 0, stream>>>(q_bf, k_bf, vt_bf, ao_bf);

  // output projection
  gqa_gemm_bf16<<<dim3(DM / 64, MR / 64), 128, 0, stream>>>(ao_bf, wo_bf, out, MR, DM, DM);
}
